// Conv3DLayer_53369263620498
// MI455X (gfx1250) — matrix operand order repaired and checked by reading, not compiled, not run on hardware
//
#include <hip/hip_runtime.h>
#include <hip/hip_bf16.h>

// ---- problem constants ----
#define BATCH   8
#define CIN     16
#define COUT    32
#define DHW     64
#define NTAPS   27            // 3*3*3
#define NUM_W   (CIN*COUT*NTAPS)   // 13824
#define NUM_P   (NUM_W + COUT)     // 13856
#define KSTEPS  14            // 28 taps (padded) / 2 taps per 16x16x32 WMMA
// A-fragments in workspace: [b][ct(2)][kstep(14)][lane(32)][16 bf16]
#define AFRAG_ELEMS (BATCH*2*KSTEPS*32*16)   // 114688 bf16
#define AFRAG_BYTES (AFRAG_ELEMS*2)          // 229376 B
#define BIAS_OFF    AFRAG_BYTES              // fp32 biases after, 8*32*4 = 1 KB

typedef __attribute__((ext_vector_type(16))) __bf16 v16bf;
typedef __attribute__((ext_vector_type(8)))  float  v8f;

__device__ __forceinline__ unsigned short f2bf(float f) {
  unsigned u = __float_as_uint(f);
  u += 0x7FFFu + ((u >> 16) & 1u);     // round-to-nearest-even
  return (unsigned short)(u >> 16);
}

// ---------------------------------------------------------------------------
// Kernel 1: hypernetwork -> per-batch conv weights (bf16, WMMA A-fragment
// order, K = tap*16 + cin, padded to 28 taps) + fp32 biases.
// ---------------------------------------------------------------------------
__global__ void __launch_bounds__(256)
hyper_params_kernel(const float* __restrict__ feat,
                    const float* __restrict__ fc0w, const float* __restrict__ fc0b,
                    const float* __restrict__ a0,
                    const float* __restrict__ fc1w, const float* __restrict__ fc1b,
                    const float* __restrict__ a1,
                    const float* __restrict__ wgw,  const float* __restrict__ wgb,
                    unsigned short* __restrict__ afrag, float* __restrict__ biasf)
{
  const int b   = blockIdx.x;
  const int tid = threadIdx.x;
  __shared__ float h0s[16];
  __shared__ float h1s[8];

  if (tid < 16) {
    float s = fc0b[tid];
    for (int j = 0; j < 10; ++j) s += feat[b*10 + j] * fc0w[tid*10 + j];
    h0s[tid] = (s >= 0.f) ? s : a0[0] * s;
  }
  __syncthreads();
  if (tid < 8) {
    float s = fc1b[tid];
    for (int j = 0; j < 16; ++j) s += h0s[j] * fc1w[tid*16 + j];
    h1s[tid] = (s >= 0.f) ? s : a1[0] * s;
  }
  __syncthreads();

  // zero the padded tap 27 (kstep 13, hi-half of K): lanes 16..31, all e, both cout tiles
  for (int i = tid; i < 512; i += 256) {
    int ct = i >> 8, rem = i & 255;
    int lane = 16 + (rem >> 4), e = rem & 15;
    afrag[(((b*2 + ct)*KSTEPS + 13)*32 + lane)*16 + e] = 0;
  }

  float h[8];
  for (int j = 0; j < 8; ++j) h[j] = h1s[j];

  for (int p = tid; p < NUM_P; p += 256) {
    float v = wgb[p];
    for (int j = 0; j < 8; ++j) v += h[j] * wgw[p*8 + j];
    if (p < NUM_W) {
      // reference layout: p = ((cout*16 + cin)*27 + tap)
      int cout = p / (CIN*NTAPS);
      int r    = p - cout*(CIN*NTAPS);
      int cin  = r / NTAPS;
      int tap  = r - cin*NTAPS;
      // scatter into 16-bit A-fragment, contiguous k order (16x32, K = tap*16+cin):
      //  lanes 0-15 hold M=row, K{0..15}; lanes 16-31 hold K{16..31}
      int ct  = cout >> 4, m = cout & 15;
      int tp  = tap >> 1,  thi = tap & 1;
      int lane = m + thi*16;
      int e    = cin;
      afrag[(((b*2 + ct)*KSTEPS + tp)*32 + lane)*16 + e] = f2bf(v);
    } else {
      biasf[b*COUT + (p - NUM_W)] = v;
    }
  }
}

// ---------------------------------------------------------------------------
// Kernel 2: implicit-GEMM 3D conv. Block = 4 waves, handles one (b,z,y) row.
// LDS slab: [dz(3)][dy(3)][xi(66)][cin(16)] bf16 -> B-fragment = one 32B
// contiguous read per lane. Each wave: 16 x-positions x 32 couts,
// 14 K-steps x 2 v_wmma_f32_16x16x32_bf16.
// ---------------------------------------------------------------------------
#define XTW 66
#define LDS_ELEMS (3*3*XTW*CIN)   // 9504 bf16 = 19008 B

__global__ void __launch_bounds__(128)
hyperconv3d_kernel(const float* __restrict__ x,
                   const unsigned short* __restrict__ afrag,
                   const float* __restrict__ biasf,
                   float* __restrict__ out)
{
  __shared__ __align__(32) unsigned short sx[LDS_ELEMS];
  const int b = blockIdx.z;
  const int z = blockIdx.x;
  const int y = blockIdx.y;
  const int tid = threadIdx.x;

  // stage halo slab (coalesced along x), fp32 -> bf16
  for (int idx = tid; idx < LDS_ELEMS; idx += 128) {
    int xi = idx % XTW;
    int r  = idx / XTW;
    int dy = r % 3; r /= 3;
    int dz = r % 3;
    int cin = r / 3;
    int gz = z + dz - 1, gy = y + dy - 1, gx = xi - 1;
    float v = 0.f;
    if (((unsigned)gz < 64u) & ((unsigned)gy < 64u) & ((unsigned)gx < 64u))
      v = x[(((b*CIN + cin)*DHW + gz)*DHW + gy)*DHW + gx];
    sx[((dz*3 + dy)*XTW + xi)*CIN + cin] = f2bf(v);
  }
  __syncthreads();

  const int wv   = tid >> 5;
  const int lane = tid & 31;
  const int n    = lane & 15;   // output column (x within wave tile)
  const int half = lane >> 4;   // which tap of the K-pair this lane's B data covers

  v8f acc0 = {0.f,0.f,0.f,0.f,0.f,0.f,0.f,0.f};
  v8f acc1 = acc0;

  const unsigned short* a0p = afrag + (((b*2 + 0)*KSTEPS)*32 + lane)*16;
  const unsigned short* a1p = afrag + (((b*2 + 1)*KSTEPS)*32 + lane)*16;

  for (int tp = 0; tp < KSTEPS; ++tp) {
    int t = 2*tp + half;
    if (t > 26) t = 0;               // padded tap: weights are zero, any valid addr
    int dz = t / 9, rr = t - dz*9;
    int dy = rr / 3, dx = rr - dy*3;
    int xi = wv*16 + n + dx;

    const v16bf bfrag = *(const v16bf*)&sx[((dz*3 + dy)*XTW + xi)*CIN];
    const v16bf af0   = *(const v16bf*)(a0p + tp*32*16);
    const v16bf af1   = *(const v16bf*)(a1p + tp*32*16);

    acc0 = __builtin_amdgcn_wmma_f32_16x16x32_bf16(false, af0, false, bfrag,
                                                   (short)0, acc0, false, false);
    acc1 = __builtin_amdgcn_wmma_f32_16x16x32_bf16(false, af1, false, bfrag,
                                                   (short)0, acc1, false, false);
  }

  // C layout: VGPR r -> M = r + 8*half, column N = lane%16
  const int gx = wv*16 + n;
  for (int r = 0; r < 8; ++r) {
    int m  = r + half*8;
    int c0 = m, c1 = m + 16;
    out[(((b*COUT + c0)*DHW + z)*DHW + y)*DHW + gx] = acc0[r] + biasf[b*COUT + c0];
    out[(((b*COUT + c1)*DHW + z)*DHW + y)*DHW + gx] = acc1[r] + biasf[b*COUT + c1];
  }
}

// ---------------------------------------------------------------------------
extern "C" void kernel_launch(void* const* d_in, const int* in_sizes, int n_in,
                              void* d_out, int out_size, void* d_ws, size_t ws_size,
                              hipStream_t stream) {
  const float* x     = (const float*)d_in[0];
  const float* feat  = (const float*)d_in[1];
  const float* fc0w  = (const float*)d_in[2];
  const float* fc0b  = (const float*)d_in[3];
  const float* a0    = (const float*)d_in[4];
  const float* fc1w  = (const float*)d_in[5];
  const float* fc1b  = (const float*)d_in[6];
  const float* a1    = (const float*)d_in[7];
  const float* wgw   = (const float*)d_in[8];
  const float* wgb   = (const float*)d_in[9];

  unsigned short* afrag = (unsigned short*)d_ws;
  float* biasf          = (float*)((char*)d_ws + BIAS_OFF);
  float* out            = (float*)d_out;

  hyper_params_kernel<<<BATCH, 256, 0, stream>>>(feat, fc0w, fc0b, a0,
                                                 fc1w, fc1b, a1, wgw, wgb,
                                                 afrag, biasf);
  hyperconv3d_kernel<<<dim3(DHW, DHW, BATCH), 128, 0, stream>>>(x, afrag, biasf, out);
}
